// SEGMENTATION_53317724012885
// MI455X (gfx1250) — matrix operand order repaired and checked by reading, not compiled, not run on hardware
//
#include <hip/hip_runtime.h>

#define NWG   32
#define TBLK  512
#define B_    64
#define T_    2048
#define V_    256
#define H_    512
#define G4H   2048   // 4*H
#define SLICE 16     // hidden units per WG
#define LCOL  64     // 4*SLICE local gate columns

typedef __attribute__((ext_vector_type(16))) __bf16          v16bf;
typedef __attribute__((ext_vector_type(8)))  float           v8f;
typedef __attribute__((ext_vector_type(16))) unsigned short  u16x16;
typedef __attribute__((ext_vector_type(8)))  unsigned short  u16x8;
typedef __attribute__((ext_vector_type(4)))  unsigned short  u16x4;

#if __has_builtin(__builtin_amdgcn_tensor_load_to_lds)
#define HAVE_TDM 1
typedef __attribute__((ext_vector_type(4))) unsigned int tdm_u32x4;
typedef __attribute__((ext_vector_type(8))) int          tdm_i32x8;
typedef __attribute__((ext_vector_type(4))) int          tdm_i32x4;
#endif

static __device__ inline unsigned short f2bf(float f) {
  unsigned u = __float_as_uint(f);
  return (unsigned short)((u + 0x7FFFu + ((u >> 16) & 1u)) >> 16);  // RNE
}
static __device__ inline float bf2f(unsigned short h) {
  return __uint_as_float(((unsigned)h) << 16);
}
static __device__ inline float sigm(float x) { return 1.f / (1.f + __expf(-x)); }

static __device__ inline void wait_tensorcnt0() {
#if __has_builtin(__builtin_amdgcn_s_wait_tensorcnt)
  __builtin_amdgcn_s_wait_tensorcnt(0);
#else
  asm volatile("s_wait_tensorcnt 0x0" ::: "memory");
#endif
}

// ---------------------------------------------------------------------------
// One-time per-call weight conversion fp32 -> bf16 into workspace (L2-resident)
// ---------------------------------------------------------------------------
__global__ void lstm_cvt_weights(const float* __restrict__ Wr,
                                 const float* __restrict__ Wk,
                                 unsigned short* __restrict__ WrB,
                                 unsigned short* __restrict__ WkB) {
  int i = blockIdx.x * blockDim.x + threadIdx.x;
  int stride = gridDim.x * blockDim.x;
  for (int p = i; p < H_ * G4H; p += stride) WrB[p] = f2bf(Wr[p]);
  for (int p = i; p < V_ * G4H; p += stride) WkB[p] = f2bf(Wk[p]);
}

// ---------------------------------------------------------------------------
// Grid-persistent LSTM: one global barrier per timestep, bf16 WMMA gate GEMMs,
// TDM (tensor_load_to_lds) broadcast of h_t into LDS each step.
// ---------------------------------------------------------------------------
__launch_bounds__(TBLK, 1)
__global__ void lstm_seg_kernel(const float* __restrict__ X,
                                const float* __restrict__ trb,
                                const int*   __restrict__ idx,
                                const float* __restrict__ bias,
                                const float* __restrict__ Wo,
                                const float* __restrict__ bo,
                                const unsigned short* __restrict__ WrB,
                                const unsigned short* __restrict__ WkB,
                                unsigned short* __restrict__ hbuf0,
                                unsigned short* __restrict__ hbuf1,
                                unsigned* __restrict__ bar,
                                float* __restrict__ out) {
  __shared__ __align__(16) unsigned short sH[B_ * H_];   // 64KB  h_t (bf16)
  __shared__ __align__(16) unsigned short sX[B_ * V_];   // 32KB  x_t (bf16)
  __shared__ __align__(16) float          sZ[B_ * LCOL]; // 16KB  gate pre-acts
  __shared__ float sC[B_ * SLICE];                        // 4KB   cell state
  __shared__ float sB[LCOL];                              // bias slice

  const int tid  = threadIdx.x;
  const int wg   = blockIdx.x;
  const int n0   = wg * SLICE;          // first hidden unit of our slice
  const int lane = tid & 31;
  const int wave = tid >> 5;            // 16 waves
  const int mt   = wave >> 2;           // M tile 0..3  (batch rows)
  const int nt   = wave & 3;            // N tile 0..3  == gate id (i,f,g,o)
  const int half = lane >> 4;
  const int lm   = lane & 15;

  // ---- init -------------------------------------------------------------
  for (int p = tid; p < B_ * SLICE; p += TBLK) sC[p] = 0.f;
  for (int c = tid; c < LCOL; c += TBLK)
    sB[c] = bias[(c >> 4) * H_ + n0 + (c & 15)];
  if (wg == 0) {
    for (int p = tid; p < B_ * 5; p += TBLK) {   // true_sel gather
      int bb = p / 5;
      out[B_ * 5 + p] = trb[(size_t)bb * T_ + idx[p]];
    }
  }
  // hbuf0/hbuf1 (h0 == 0) and bar[] were zeroed by hipMemsetAsync.

  unsigned short* hb[2] = {hbuf0, hbuf1};

#if HAVE_TDM
  const unsigned sH_lds =
      (unsigned)(uintptr_t)(__attribute__((address_space(3))) unsigned short*)&sH[0];
#endif

  for (int t = 0; t < T_; ++t) {
    const unsigned short* hcur = hb[t & 1];
    unsigned short*       hnxt = hb[(t + 1) & 1];

    // ---- phase A: stage h_t (TDM DMA) and x_t (vectorized cvt) into LDS --
#if HAVE_TDM
    if (tid == 0) {
      const unsigned nelem = B_ * H_;                  // 32768 bf16 elements
      unsigned long long ga = (unsigned long long)(uintptr_t)hcur;
      tdm_u32x4 g0 = {1u,                              // count=1 (user D#)
                      sH_lds,                          // lds_addr
                      (unsigned)(ga & 0xFFFFFFFFu),    // global_addr[31:0]
                      (unsigned)((ga >> 32) & 0x01FFFFFFu) | (2u << 30)}; // +type=2
      tdm_i32x8 g1 = {0, 0, 0, 0, 0, 0, 0, 0};
      g1[0] = (int)(1u << 16);                         // data_size = 2 bytes
      g1[1] = (int)((nelem & 0xFFFFu) << 16);          // tensor_dim0[15:0]
      g1[2] = (int)(((nelem >> 16) & 0xFFFFu) | (1u << 16)); // dim0 hi | tensor_dim1=1
      g1[3] = (int)((nelem & 0xFFFFu) << 16);          // tile_dim0 = 32768
      g1[5] = (int)nelem;                              // tensor_dim0_stride
      tdm_i32x4 gz = {0, 0, 0, 0};
      tdm_i32x8 gz8 = {0, 0, 0, 0, 0, 0, 0, 0};
      __builtin_amdgcn_tensor_load_to_lds(g0, g1, gz, gz, gz8, 0);
    }
#else
    for (int p = tid * 8; p < B_ * H_; p += TBLK * 8)
      *(u16x8*)&sH[p] = *(const u16x8*)&hcur[p];
#endif
    for (int p = tid; p < (B_ * V_) / 4; p += TBLK) {  // x_t: float4 -> 4x bf16
      int bb = p >> 6, v4 = (p & 63) * 4;
      float4 xv = *(const float4*)&X[((size_t)bb * T_ + t) * V_ + v4];
      u16x4 hv;
      hv[0] = f2bf(xv.x); hv[1] = f2bf(xv.y);
      hv[2] = f2bf(xv.z); hv[3] = f2bf(xv.w);
      *(u16x4*)&sX[bb * V_ + v4] = hv;
    }
    if (t + 1 < T_) {  // prefetch next timestep's activations
      for (int bb = tid; bb < B_; bb += TBLK)
        __builtin_prefetch(&X[((size_t)bb * T_ + (t + 1)) * V_], 0, 1);
    }
#if HAVE_TDM
    if (wave == 0) wait_tensorcnt0();   // issuing wave drains TENSORcnt
#endif
    __syncthreads();

    // ---- phase B: Z(mt,nt) = h @ Wr + x @ Wk + b  (bf16 WMMA) ------------
    float bv = sB[nt * 16 + lm];     // C layout: lane holds column N = lane%16
    v8f acc;
#pragma unroll
    for (int i = 0; i < 8; ++i) acc[i] = bv;

    const unsigned short* arow = &sH[(mt * 16 + lm) * H_ + half * 8];
    const unsigned short* brow = &WrB[(size_t)(half * 8) * G4H + nt * H_ + n0 + lm];
#pragma unroll
    for (int k0 = 0; k0 < H_; k0 += 32) {
      u16x8 alo = *(const u16x8*)(arow + k0);        // K = k0+8h .. +7
      u16x8 ahi = *(const u16x8*)(arow + k0 + 16);   // K = k0+16+8h ..
      u16x16 a;
#pragma unroll
      for (int i = 0; i < 8; ++i) { a[i] = alo[i]; a[i + 8] = ahi[i]; }
      u16x16 bm;  // rows k0+8h+i and k0+16+8h+i, column lm
#pragma unroll
      for (int i = 0; i < 8; ++i) { bm[i] = brow[(size_t)(k0 + i) * G4H]; bm[i + 8] = brow[(size_t)(k0 + 16 + i) * G4H]; }
      acc = __builtin_amdgcn_wmma_f32_16x16x32_bf16(
          false, __builtin_bit_cast(v16bf, a),
          false, __builtin_bit_cast(v16bf, bm),
          (short)0, acc, false, false);
    }
    const unsigned short* xrow = &sX[(mt * 16 + lm) * V_ + half * 8];
    const unsigned short* krow = &WkB[(size_t)(half * 8) * G4H + nt * H_ + n0 + lm];
#pragma unroll
    for (int k0 = 0; k0 < V_; k0 += 32) {
      u16x8 alo = *(const u16x8*)(xrow + k0);
      u16x8 ahi = *(const u16x8*)(xrow + k0 + 16);
      u16x16 a;
#pragma unroll
      for (int i = 0; i < 8; ++i) { a[i] = alo[i]; a[i + 8] = ahi[i]; }
      u16x16 bm;
#pragma unroll
      for (int i = 0; i < 8; ++i) { bm[i] = krow[(size_t)(k0 + i) * G4H]; bm[i + 8] = krow[(size_t)(k0 + 16 + i) * G4H]; }
      acc = __builtin_amdgcn_wmma_f32_16x16x32_bf16(
          false, __builtin_bit_cast(v16bf, a),
          false, __builtin_bit_cast(v16bf, bm),
          (short)0, acc, false, false);
    }
#pragma unroll
    for (int v = 0; v < 8; ++v) {    // D layout: VGPR v -> row v + 8*half
      int m = mt * 16 + v + 8 * half;
      sZ[m * LCOL + nt * 16 + lm] = acc[v];
    }
    __syncthreads();

    // ---- phase C: gate nonlinearity, update c, publish h slice -----------
    for (int p = tid; p < B_ * SLICE; p += TBLK) {
      int m = p >> 4, n = p & 15;
      float zi = sZ[m * LCOL + n];
      float zf = sZ[m * LCOL + 16 + n];
      float zg = sZ[m * LCOL + 32 + n];
      float zo = sZ[m * LCOL + 48 + n];
      float cc = sigm(zf) * sC[p] + sigm(zi) * tanhf(zg);
      float hh = sigm(zo) * tanhf(cc);
      sC[p] = cc;
      hnxt[m * H_ + n0 + n] = f2bf(hh);
    }
    __threadfence();

    // ---- device-wide sense-reversing barrier -----------------------------
    __syncthreads();
    if (tid == 0) {
      unsigned g = __hip_atomic_load(&bar[1], __ATOMIC_RELAXED,
                                     __HIP_MEMORY_SCOPE_AGENT);
      unsigned a = __hip_atomic_fetch_add(&bar[0], 1u, __ATOMIC_ACQ_REL,
                                          __HIP_MEMORY_SCOPE_AGENT);
      if (a == NWG - 1) {
        __hip_atomic_store(&bar[0], 0u, __ATOMIC_RELAXED,
                           __HIP_MEMORY_SCOPE_AGENT);
        __hip_atomic_store(&bar[1], g + 1u, __ATOMIC_RELEASE,
                           __HIP_MEMORY_SCOPE_AGENT);
      } else {
        while (__hip_atomic_load(&bar[1], __ATOMIC_ACQUIRE,
                                 __HIP_MEMORY_SCOPE_AGENT) == g)
          __builtin_amdgcn_s_sleep(2);
      }
    }
    __syncthreads();

    // ---- phase D: sigmoid head at selected timesteps (WG0) ---------------
    if (wg == 0) {
      for (int p = tid; p < B_ * 5; p += TBLK) {
        if (idx[p] == t) {
          int bb = p / 5;
          const unsigned short* hr = hnxt + bb * H_;
          float a = bo[0];
#pragma unroll 8
          for (int k = 0; k < H_; ++k) a += bf2f(hr[k]) * Wo[k];
          out[p] = sigm(a);
        }
      }
    }
  }
}

// ---------------------------------------------------------------------------
extern "C" void kernel_launch(void* const* d_in, const int* in_sizes, int n_in,
                              void* d_out, int out_size, void* d_ws,
                              size_t ws_size, hipStream_t stream) {
  const float* X   = (const float*)d_in[0];
  const float* trb = (const float*)d_in[1];
  const int*   idx = (const int*)d_in[2];
  const float* Wk  = (const float*)d_in[3];
  const float* Wr  = (const float*)d_in[4];
  const float* b   = (const float*)d_in[5];
  const float* Wo  = (const float*)d_in[6];
  const float* bo  = (const float*)d_in[7];
  float* out = (float*)d_out;

  char* ws = (char*)d_ws;
  const size_t HBYT = (size_t)B_ * H_ * sizeof(unsigned short);  // 64KB
  unsigned*       bar   = (unsigned*)ws;                           // 256 B
  unsigned short* hbuf0 = (unsigned short*)(ws + 256);
  unsigned short* hbuf1 = (unsigned short*)(ws + 256 + HBYT);
  unsigned short* WrB   = (unsigned short*)(ws + 256 + 2 * HBYT);
  unsigned short* WkB =
      (unsigned short*)(ws + 256 + 2 * HBYT + (size_t)H_ * G4H * 2);

  // zero barrier state + h0 ping-pong buffers (graph-capture legal)
  (void)hipMemsetAsync(d_ws, 0, 256 + 2 * HBYT, stream);
  lstm_cvt_weights<<<256, 256, 0, stream>>>(Wr, Wk, WrB, WkB);
  lstm_seg_kernel<<<NWG, TBLK, 0, stream>>>(X, trb, idx, b, Wo, bo, WrB, WkB,
                                            hbuf0, hbuf1, bar, out);
}
